// RelativePositionMultiheadAttention_24816321036395
// MI455X (gfx1250) — compile-verified, matrix operand order checked by reading, not run on hardware
//
#include <hip/hip_runtime.h>
#include <stdint.h>

// ---------------- CDNA5 WMMA types ----------------
typedef __attribute__((ext_vector_type(16))) __bf16        v16bf;
typedef __attribute__((ext_vector_type(8)))  float         v8f;
typedef __attribute__((ext_vector_type(4)))  unsigned int  u32x4;
typedef __attribute__((ext_vector_type(8)))  int           i32x8;
typedef __attribute__((ext_vector_type(4)))  int           i32x4;

// Builtin prototype wants int* in AS1 (global) / AS3 (LDS), non-const.
typedef __attribute__((address_space(1))) int* as1i;
typedef __attribute__((address_space(3))) int* as3i;

union FragBF {
    v16bf v;
    u32x4 q[2];
    unsigned short u[16];
};

__device__ __forceinline__ unsigned short f2bf(float f) {
    unsigned u = __float_as_uint(f);
    unsigned r = u + 0x7FFFu + ((u >> 16) & 1u);   // round-to-nearest-even
    return (unsigned short)(r >> 16);
}
__device__ __forceinline__ float bf2f(unsigned short h) {
    return __uint_as_float(((unsigned)h) << 16);
}

// Load a 16x32 bf16 fragment (A or B operand of V_WMMA_F32_16X16X32_BF16).
// Per ISA layout: lane L -> row/col (L&15); two contiguous 8-element (16B)
// chunks at K = (L>>4)*8 and K = (L>>4)*8 + 16.
__device__ __forceinline__ FragBF load_frag(const unsigned short* __restrict__ src,
                                            size_t row0, size_t col0,
                                            size_t rowStride, int lane) {
    FragBF f;
    size_t row = row0 + (size_t)(lane & 15);
    size_t c0  = col0 + (size_t)((lane >> 4) << 3);
    const unsigned short* p = src + row * rowStride + c0;
    f.q[0] = *(const u32x4*)(p);
    f.q[1] = *(const u32x4*)(p + 16);
    return f;
}

__device__ __forceinline__ v8f wmma_bf16(const FragBF& a, const FragBF& b, v8f c) {
    return __builtin_amdgcn_wmma_f32_16x16x32_bf16(false, a.v, false, b.v,
                                                   (short)0, c, false, false);
}

// ---------------- CDNA5 async / TDM helpers (guarded) ----------------
#if __has_builtin(__builtin_amdgcn_global_load_async_to_lds_b32)
#define HAVE_ASYNC_LDS 1
#endif
#if __has_builtin(__builtin_amdgcn_tensor_load_to_lds)
#define HAVE_TDM 1
#endif

__device__ __forceinline__ void wait_asynccnt0() {
#if __has_builtin(__builtin_amdgcn_s_wait_asynccnt)
    __builtin_amdgcn_s_wait_asynccnt(0);
#else
    asm volatile("s_wait_asynccnt 0x0" ::: "memory");
#endif
}
__device__ __forceinline__ void wait_tensorcnt0() {
#if __has_builtin(__builtin_amdgcn_s_wait_tensorcnt)
    __builtin_amdgcn_s_wait_tensorcnt(0);
#else
    asm volatile("s_wait_tensorcnt 0x0" ::: "memory");
#endif
}

#ifdef HAVE_TDM
// One contiguous run of `count` f32 elements: 1D tile, data_size=4B.
// D# packing per CDNA5 ISA.
__device__ __forceinline__ void tdm_load_f32_1d(void* lds_dst, const void* gsrc,
                                                unsigned count) {
    unsigned long long ga = (unsigned long long)(uintptr_t)gsrc;
    unsigned lad = (unsigned)(uintptr_t)lds_dst;   // low 32 bits = LDS byte offset
    u32x4 g0;
    g0[0] = 1u;                                              // count=1 valid descriptor
    g0[1] = lad;                                             // lds_addr
    g0[2] = (unsigned)ga;                                    // global_addr[31:0]
    g0[3] = (unsigned)((ga >> 32) & 0x01FFFFFFu) | (2u << 30); // [56:32] + type=2
    i32x8 g1;
    g1[0] = (int)(2u << 16);                 // data_size=2 (4 bytes)
    g1[1] = (int)((count & 0xFFFFu) << 16);  // tensor_dim0[15:0]
    g1[2] = (int)(1u << 16);                 // tensor_dim0[31:16]=0, tensor_dim1=1
    g1[3] = (int)((count & 0xFFFFu) << 16);  // tile_dim0
    g1[4] = 1;                               // tile_dim1=1, tile_dim2=0
    g1[5] = (int)count;                      // tensor_dim0_stride[31:0]
    g1[6] = 0;
    g1[7] = 0;
    i32x4 z = {};
#if __has_include(<hip/amd_detail/amd_gfx1250_TDM.h>)
    i32x8 z8 = {};
    __builtin_amdgcn_tensor_load_to_lds(g0, g1, z, z, z8, 0);   // 6-arg toolchain
#else
    __builtin_amdgcn_tensor_load_to_lds(g0, g1, z, z, 0);       // 5-arg ROCm 7.2
#endif
}
#endif

// ---------------- problem constants ----------------
#define TT   1024
#define BB   4
#define EE   1024
#define HH   16
#define DD   64
#define BH   64          // BB*HH
#define F3   3072        // 3*EE
#define MM   4096        // TT*BB

// ---------------- f32 -> bf16 convert ----------------
__global__ void __launch_bounds__(256) k_f32_to_bf16(const float* __restrict__ src,
                                                     unsigned short* __restrict__ dst,
                                                     int n) {
    int i = blockIdx.x * 256 + threadIdx.x;
    if (i < n) dst[i] = f2bf(src[i]);
}

// ---------------- GEMM1: qkv = query @ Wᵀ + bias, scatter to heads ----------------
// 64x64 register tile per wave: 16 WMMAs per 8 fragment loads (32 FLOP/B).
__global__ void __launch_bounds__(256) k_gemm_qkv(const unsigned short* __restrict__ Abf,
                                                  const unsigned short* __restrict__ Wbf,
                                                  const float* __restrict__ bias,
                                                  unsigned short* __restrict__ qh,   // [64][1024][64]
                                                  unsigned short* __restrict__ kh,   // [64][1024][64]
                                                  unsigned short* __restrict__ vhT)  // [64][64][1024]
{
    const int lane = threadIdx.x & 31;
    const int wave = blockIdx.x * 8 + (threadIdx.x >> 5);
    const int tilesN = F3 / 64;
    const int tm = wave / tilesN;
    const int tn = wave % tilesN;

    v8f acc[16] = {};
    for (int k = 0; k < EE; k += 32) {
        FragBF af[4], bf[4];
#pragma unroll
        for (int i = 0; i < 4; ++i)
            af[i] = load_frag(Abf, (size_t)tm * 64 + i * 16, k, EE, lane);
#pragma unroll
        for (int j = 0; j < 4; ++j)
            bf[j] = load_frag(Wbf, (size_t)tn * 64 + j * 16, k, EE, lane);
        if (k + 32 < EE) {  // CDNA5 global_prefetch path
            __builtin_prefetch(Abf + ((size_t)tm * 64 + (lane & 15)) * EE + k + 32, 0, 1);
            __builtin_prefetch(Wbf + ((size_t)tn * 64 + (lane & 15)) * EE + k + 32, 0, 1);
        }
#pragma unroll
        for (int i = 0; i < 4; ++i)
#pragma unroll
            for (int j = 0; j < 4; ++j)
                acc[i * 4 + j] = wmma_bf16(af[i], bf[j], acc[i * 4 + j]);
    }

#pragma unroll
    for (int j = 0; j < 4; ++j) {
        const int f  = tn * 64 + j * 16 + (lane & 15);
        const float bv = bias[f];
        const int c3 = f >> 10;     // 0=q 1=k 2=v  (uniform: 64-aligned tile)
        const int e  = f & 1023;
        const int h  = e >> 6;
        const int dcol = e & 63;
#pragma unroll
        for (int i = 0; i < 4; ++i) {
#pragma unroll
            for (int r = 0; r < 8; ++r) {
                int m = tm * 64 + i * 16 + r + ((lane >> 4) << 3);
                int t = m >> 2, bidx = m & 3;
                int nn = bidx * HH + h;
                float val = acc[i * 4 + j][r] + bv;
                if (c3 == 0)      qh[(((size_t)nn * TT + t) << 6) + dcol] = f2bf(val * 0.125f);
                else if (c3 == 1) kh[(((size_t)nn * TT + t) << 6) + dcol] = f2bf(val);
                else              vhT[(((size_t)nn * DD + dcol) << 10) + t] = f2bf(val);
            }
        }
    }
}

// ---------------- rb[n][t][i] = dot(q_scaled[n,t,:], rel_k_table[i,:]) ----------------
__global__ void __launch_bounds__(256) k_rb(const unsigned short* __restrict__ qh,
                                            const float* __restrict__ relk,  // [33][64]
                                            float* __restrict__ rb)          // [64][1024][33]
{
    int idx = blockIdx.x * 256 + threadIdx.x;   // 65536 = 64*1024
    const unsigned short* qp = qh + ((size_t)idx << 6);
    float qv[DD];
#pragma unroll
    for (int d = 0; d < DD; ++d) qv[d] = bf2f(qp[d]);
    float* rbo = rb + (size_t)idx * 33;
    for (int i = 0; i < 33; ++i) {
        const float* rk = relk + i * DD;
        float s = 0.f;
#pragma unroll
        for (int d = 0; d < DD; ++d) s += qv[d] * rk[d];
        rbo[i] = s;
    }
}

// ---------------- attention: scores + softmax + P·V + rel_v ----------------
// grid: 64 heads * 16 q-blocks ; block: 128 threads (4 waves, 16 q-rows each)
__global__ void __launch_bounds__(128) k_attn(const unsigned short* __restrict__ qh,
                                              const unsigned short* __restrict__ kh,
                                              const unsigned short* __restrict__ vhT,
                                              const float* __restrict__ rb,
                                              const float* __restrict__ relv,   // [33][64]
                                              unsigned short* __restrict__ attn_bf) // [4096][1024]
{
    extern __shared__ float smem[];
    float* scores = smem;                    // [64][1024]
    float* rbs    = scores + 64 * 1024;      // [64][33]
    float* rvs    = rbs    + 64 * 33;        // [33][64]
    float* relbuf = rvs    + 33 * 64;        // [64][64]
    float* rstats = relbuf + 64 * 64;        // [64][3]  (l, s0, s32)

    const int tid  = threadIdx.x;
    const int lane = tid & 31;
    const int w    = tid >> 5;
    const int n    = blockIdx.x >> 4;            // head-batch index 0..63
    const int q0g  = (blockIdx.x & 15) * 64;     // block's first q row

    // ---- stage rb rows (TDM) + rel_v table (async-LDS) ----
#ifdef HAVE_TDM
    if (tid < 32) {                          // one wave issues the DMA
        tdm_load_f32_1d(rbs, rb + ((size_t)n * TT + q0g) * 33, 64 * 33);
        wait_tensorcnt0();
    }
#else
    for (int i = tid; i < 64 * 33; i += 128) {
        int rr = i / 33, ii = i - rr * 33;
        rbs[i] = rb[((size_t)n * TT + q0g + rr) * 33 + ii];
    }
#endif
#ifdef HAVE_ASYNC_LDS
    for (int i = tid; i < 33 * 64; i += 128) {
        __builtin_amdgcn_global_load_async_to_lds_b32(
            (as1i)(uintptr_t)(relv + i),
            (as3i)(unsigned)(uintptr_t)(rvs + i), 0, 0);
    }
    wait_asynccnt0();
#else
    for (int i = tid; i < 33 * 64; i += 128) rvs[i] = relv[i];
#endif
    __syncthreads();

    const int qw = q0g + w * 16;                 // this wave's first q row
    const unsigned short* qbase = qh  + ((size_t)n * TT) * DD;
    const unsigned short* kbase = kh  + ((size_t)n * TT) * DD;
    const unsigned short* vbase = vhT + ((size_t)n * DD) * TT;

    FragBF aq0 = load_frag(qbase, qw, 0,  DD, lane);
    FragBF aq1 = load_frag(qbase, qw, 32, DD, lane);

    // ---- phase 1: scores = q·kᵀ + rel_k bias ----
    for (int kt = 0; kt < TT / 16; ++kt) {
        int k0 = kt * 16;
        FragBF b0 = load_frag(kbase, k0, 0,  DD, lane);
        FragBF b1 = load_frag(kbase, k0, 32, DD, lane);
        v8f c = {};
        c = wmma_bf16(aq0, b0, c);
        c = wmma_bf16(aq1, b1, c);
        int kk = k0 + (lane & 15);
#pragma unroll
        for (int r = 0; r < 8; ++r) {
            int M = r + ((lane >> 4) << 3);
            int qg = qw + M;
            int dist = kk - qg;
            int ci = dist < -16 ? 0 : (dist > 16 ? 32 : dist + 16);
            int srow = w * 16 + M;
            scores[srow * 1024 + kk] = c[r] + rbs[srow * 33 + ci];
        }
    }

    // ---- phase 2: row softmax stats, exp in place (2 lanes per row) ----
    {
        int r = lane & 15, half = lane >> 4;
        int srow = w * 16 + r;
        int qg = qw + r;
        float* row = scores + srow * 1024 + half * 512;
        float mx = -1e30f;
        for (int j = 0; j < 512; ++j) mx = fmaxf(mx, row[j]);
        mx = fmaxf(mx, __shfl_xor(mx, 16, 32));
        float lsum = 0.f, s0 = 0.f, s32v = 0.f;
        int kb = half * 512;
        for (int j = 0; j < 512; ++j) {
            float ev = __expf(row[j] - mx);
            row[j] = ev;
            lsum += ev;
            int kk = kb + j;
            if (kk <= qg - 16)      s0   += ev;
            else if (kk >= qg + 16) s32v += ev;
        }
        lsum += __shfl_xor(lsum, 16, 32);
        s0   += __shfl_xor(s0,   16, 32);
        s32v += __shfl_xor(s32v, 16, 32);
        if (half == 0) {
            rstats[srow * 3 + 0] = lsum;
            rstats[srow * 3 + 1] = s0;
            rstats[srow * 3 + 2] = s32v;
        }
    }

    // ---- phase 3: unnormalized rel_v term per row (33 classes) ----
    for (int r = 0; r < 16; ++r) {
        int srow = w * 16 + r;
        int qg = qw + r;
        float s0   = rstats[srow * 3 + 1];
        float s32v = rstats[srow * 3 + 2];
        int d1 = lane, d2 = lane + 32;
        float a1 = s0 * rvs[d1] + s32v * rvs[32 * 64 + d1];
        float a2 = s0 * rvs[d2] + s32v * rvs[32 * 64 + d2];
        for (int i = 1; i <= 31; ++i) {
            int kk = qg - 16 + i;
            if (kk >= 0 && kk < TT) {
                float p = scores[srow * 1024 + kk];
                a1 += p * rvs[i * 64 + d1];
                a2 += p * rvs[i * 64 + d2];
            }
        }
        relbuf[srow * 64 + d1] = a1;
        relbuf[srow * 64 + d2] = a2;
    }

    // ---- phase 4: out = (E·V + rel) / l  via WMMA over K=32 chunks ----
    v8f acc[4] = {};
    for (int kt = 0; kt < TT / 32; ++kt) {
        int kk0 = kt * 32;
        FragBF a;
        {
            int srow = w * 16 + (lane & 15);
            const float* p = scores + srow * 1024 + kk0 + ((lane >> 4) << 3);
#pragma unroll
            for (int j = 0; j < 8; ++j) {
                a.u[j]     = f2bf(p[j]);
                a.u[8 + j] = f2bf(p[16 + j]);
            }
        }
#pragma unroll
        for (int dc = 0; dc < 4; ++dc) {
            int dcol = dc * 16 + (lane & 15);
            const unsigned short* vp = vbase + (size_t)dcol * TT + kk0 + ((lane >> 4) << 3);
            FragBF b;
            b.q[0] = *(const u32x4*)vp;
            b.q[1] = *(const u32x4*)(vp + 16);
            acc[dc] = wmma_bf16(a, b, acc[dc]);
        }
    }

    const int bidx = n >> 4, h = n & 15;
#pragma unroll
    for (int dc = 0; dc < 4; ++dc) {
        int dcol = dc * 16 + (lane & 15);
#pragma unroll
        for (int r = 0; r < 8; ++r) {
            int M = r + ((lane >> 4) << 3);
            int srow = w * 16 + M;
            int qg = qw + M;
            float linv = 1.0f / rstats[srow * 3 + 0];
            float val = (acc[dc][r] + relbuf[srow * 64 + dcol]) * linv;
            size_t m = (size_t)qg * BB + bidx;
            attn_bf[m * EE + (size_t)h * DD + dcol] = f2bf(val);
        }
    }
}

// ---------------- GEMM2: out = attn @ out_wᵀ + out_b (f32 output) ----------------
__global__ void __launch_bounds__(256) k_gemm_out(const unsigned short* __restrict__ Abf,
                                                  const unsigned short* __restrict__ Wbf,
                                                  const float* __restrict__ bias,
                                                  float* __restrict__ out)
{
    const int lane = threadIdx.x & 31;
    const int wave = blockIdx.x * 8 + (threadIdx.x >> 5);
    const int tilesN = EE / 64;
    const int tm = wave / tilesN;
    const int tn = wave % tilesN;

    v8f acc[16] = {};
    for (int k = 0; k < EE; k += 32) {
        FragBF af[4], bf[4];
#pragma unroll
        for (int i = 0; i < 4; ++i)
            af[i] = load_frag(Abf, (size_t)tm * 64 + i * 16, k, EE, lane);
#pragma unroll
        for (int j = 0; j < 4; ++j)
            bf[j] = load_frag(Wbf, (size_t)tn * 64 + j * 16, k, EE, lane);
        if (k + 32 < EE) {
            __builtin_prefetch(Abf + ((size_t)tm * 64 + (lane & 15)) * EE + k + 32, 0, 1);
            __builtin_prefetch(Wbf + ((size_t)tn * 64 + (lane & 15)) * EE + k + 32, 0, 1);
        }
#pragma unroll
        for (int i = 0; i < 4; ++i)
#pragma unroll
            for (int j = 0; j < 4; ++j)
                acc[i * 4 + j] = wmma_bf16(af[i], bf[j], acc[i * 4 + j]);
    }

#pragma unroll
    for (int j = 0; j < 4; ++j) {
        int f = tn * 64 + j * 16 + (lane & 15);
        float bv = bias[f];
#pragma unroll
        for (int i = 0; i < 4; ++i)
#pragma unroll
            for (int r = 0; r < 8; ++r) {
                int m = tm * 64 + i * 16 + r + ((lane >> 4) << 3);
                out[(size_t)m * EE + f] = acc[i * 4 + j][r] + bv;
            }
    }
}

// ---------------- host launch ----------------
extern "C" void kernel_launch(void* const* d_in, const int* in_sizes, int n_in,
                              void* d_out, int out_size, void* d_ws, size_t ws_size,
                              hipStream_t stream) {
    const float* query = (const float*)d_in[0];
    const float* ipw   = (const float*)d_in[1];
    const float* ipb   = (const float*)d_in[2];
    const float* relk  = (const float*)d_in[3];
    const float* relv  = (const float*)d_in[4];
    const float* outw  = (const float*)d_in[5];
    const float* outb  = (const float*)d_in[6];
    float* out = (float*)d_out;

    char* ws = (char*)d_ws;
    size_t off = 0;
    auto alloc = [&](size_t bytes) -> void* {
        void* p = ws + off;
        off += (bytes + 255) & ~(size_t)255;
        return p;
    };
    unsigned short* q_bf  = (unsigned short*)alloc((size_t)MM * EE * 2); // reused as attn_bf
    unsigned short* w_bf  = (unsigned short*)alloc((size_t)F3 * EE * 2);
    unsigned short* ow_bf = (unsigned short*)alloc((size_t)EE * EE * 2);
    unsigned short* qhB   = (unsigned short*)alloc((size_t)BH * TT * DD * 2);
    unsigned short* khB   = (unsigned short*)alloc((size_t)BH * TT * DD * 2);
    unsigned short* vhT   = (unsigned short*)alloc((size_t)BH * DD * TT * 2);
    float* rbB            = (float*)alloc((size_t)BH * TT * 33 * 4);

    // converts
    {
        int n1 = MM * EE;
        k_f32_to_bf16<<<(n1 + 255) / 256, 256, 0, stream>>>(query, q_bf, n1);
        int n2 = F3 * EE;
        k_f32_to_bf16<<<(n2 + 255) / 256, 256, 0, stream>>>(ipw, w_bf, n2);
        int n3 = EE * EE;
        k_f32_to_bf16<<<(n3 + 255) / 256, 256, 0, stream>>>(outw, ow_bf, n3);
    }

    // qkv projection + head scatter (64x64 tile per wave)
    k_gemm_qkv<<<(MM / 64) * (F3 / 64) / 8, 256, 0, stream>>>(q_bf, w_bf, ipb, qhB, khB, vhT);

    // relative-key bias table
    k_rb<<<(BH * TT) / 256, 256, 0, stream>>>(qhB, relk, rbB);

    // attention (4 waves per block; big dynamic LDS: scores+tables ~289 KB)
    size_t smem = (size_t)(64 * 1024 + 64 * 33 + 33 * 64 + 64 * 64 + 64 * 3) * sizeof(float);
    k_attn<<<BH * (TT / 64), 128, smem, stream>>>(qhB, khB, vhT, rbB, relv, q_bf);

    // output projection
    k_gemm_out<<<(MM / 64) * (EE / 64) / 8, 256, 0, stream>>>(q_bf, ow_bf, outb, out);
}
